// DynamicGNNEncoder_59433757442507
// MI455X (gfx1250) — compile-verified, matrix operand order checked by reading, not run on hardware
//
#include <hip/hip_runtime.h>
#include <hip/hip_bf16.h>
#include <stdint.h>
#include <stddef.h>

typedef __bf16 bf16;
typedef __attribute__((ext_vector_type(16))) __bf16 v16bf;
typedef __attribute__((ext_vector_type(8)))  float  v8f;
typedef __attribute__((ext_vector_type(4)))  unsigned int u32x4;
typedef __attribute__((ext_vector_type(2)))  unsigned int u32x2;
typedef __attribute__((ext_vector_type(4)))  float f32x4;

#define B_     32
#define N_     512
#define DPPL   2048
#define DANS   1024
#define DVH    300
#define HID_   512
#define KTOP   10
#define EPSV   0.7f
#define NLAYER 3
#define STEPS_ 36

// ---------------- small prep kernels ----------------

// u[j] = sum_k W_vh[k] * W_dh[2048+k, j] ; c[j] = sum_k b_vh[k] * W_dh[2048+k, j]
__global__ void prep_uc_kernel(const float* __restrict__ W_vh, const float* __restrict__ b_vh,
                               const float* __restrict__ W_dh,
                               float* __restrict__ u, float* __restrict__ c) {
  int j = threadIdx.x;  // 512
  float su = 0.f, sc = 0.f;
  for (int k = 0; k < DVH; ++k) {
    float w = W_dh[(size_t)(DPPL + k) * HID_ + j];
    su += W_vh[k] * w;
    sc += b_vh[k] * w;
  }
  u[j] = su;
  c[j] = sc;
}

// bias[b,j] = answer[b] @ W_dh_ans[:,j] + b_dh[j] + c[j]
__global__ void prep_bias_kernel(const float* __restrict__ answer, const float* __restrict__ W_dh,
                                 const float* __restrict__ b_dh, const float* __restrict__ c,
                                 float* __restrict__ bias) {
  int b = blockIdx.x, j = threadIdx.x;
  float s = b_dh[j] + c[j];
  const float* arow = answer + (size_t)b * DANS;
  for (int k = 0; k < DANS; ++k)
    s += arow[k] * W_dh[(size_t)(DPPL + DVH + k) * HID_ + j];
  bias[b * HID_ + j] = s;
}

// dst[z][n][k] = (bf16) src[z][k][n]  -- 32x32 LDS tiled transpose + convert
__global__ void cvt_transpose_kernel(const float* __restrict__ src, bf16* __restrict__ dst,
                                     int K, int Ncols) {
  __shared__ float tile[32][33];
  const float* s = src + (size_t)blockIdx.z * K * Ncols;
  bf16*        d = dst + (size_t)blockIdx.z * K * Ncols;
  int k0 = blockIdx.x * 32, n0 = blockIdx.y * 32;
  int tx = threadIdx.x & 31, ty = threadIdx.x >> 5;  // 32 x 8
#pragma unroll
  for (int dy = 0; dy < 32; dy += 8)
    tile[ty + dy][tx] = s[(size_t)(k0 + ty + dy) * Ncols + n0 + tx];
  __syncthreads();
#pragma unroll
  for (int dy = 0; dy < 32; dy += 8)
    d[(size_t)(n0 + ty + dy) * K + k0 + tx] = (bf16)tile[tx][ty + dy];
}

// ---------------- WMMA GEMM (double-buffered, software pipelined) ----------------
// B operand is ALWAYS provided K-major: Bt[n][k] (= B^T row-major), ld = Kk.
// So staging both A and B tiles is a contiguous 16-byte row copy -- no LDS transposes.

#define BM  128
#define BN  128
#define BK  32
#define LDT 40  // padded LDS row stride in halves (80 bytes, keeps 16B alignment)

union FragBF { u32x4 u[2]; v16bf v; };
union Pack8  { u32x4 u; bf16 h[8]; };
union Pack4  { u32x2 u; bf16 h[4]; };

// AMODE: 0 = A f32 row-major (convert while staging), 1 = A bf16 row-major
// EMODE: 0 = xw epilogue: relu(acc + bias[b,j] + vh[r]*u[j]) * w_topo[j] -> f32 row-major
//        2 = raw f32 row-major
//        3 = raw bf16 row-major
//        4 = relu(acc + bias[j]) -> f32 row-major
//        5 = relu(acc + bias[j]) -> bf16 TRANSPOSED ht[b][n][m%512], packed 8x-bf16 stores
template <int AMODE, int EMODE>
__global__ __launch_bounds__(256)
void gemm_wmma_kernel(const void* __restrict__ Ag, const void* __restrict__ Bg,
                      void* __restrict__ Cg, int M, int Nn, int Kk,
                      long sA, long sB, long sC,
                      const float* __restrict__ ep0, const float* __restrict__ ep1,
                      const float* __restrict__ ep2, const float* __restrict__ ep3) {
  __shared__ __align__(16) bf16 As[2][BM * LDT];
  __shared__ __align__(16) bf16 Bs[2][BN * LDT];

  const int tid = threadIdx.x;
  const int bm = blockIdx.x, bn = blockIdx.y, bz = blockIdx.z;

  const float* A32 = (const float*)Ag + (size_t)bz * sA;
  const bf16*  A16 = (const bf16*)Ag + (size_t)bz * sA;
  const bf16*  B16 = (const bf16*)Bg + (size_t)bz * sB;

  const v8f vzero = {0.f, 0.f, 0.f, 0.f, 0.f, 0.f, 0.f, 0.f};
  v8f acc[4][2];
#pragma unroll
  for (int i = 0; i < 4; ++i)
#pragma unroll
    for (int j = 0; j < 2; ++j) acc[i][j] = vzero;

  const int lane   = tid & 31;
  const int waveId = tid >> 5;
  const int wm     = (waveId >> 2) * 64;  // 2 waves along M
  const int wn     = (waveId & 3) * 32;   // 4 waves along N
  const int lanelo = lane & 15;
  const int klo    = (lane >> 4) * 8;     // per-lane K offset (halves), per ISA layout

  // fixed staging indices (no data-dependent bounds -> no exec-mask branches)
  const int ra = tid >> 3, c4 = (tid & 7) * 4;  // AMODE0: 128x32 f32, 4 rows/thread
  const int rb = tid >> 2, c8 = (tid & 3) * 8;  // bf16 row copy: 2 rows/thread

  f32x4 fa[4];   // staged f32 A
  u32x4 a16[2];  // staged bf16 A
  u32x4 breg[2]; // staged B

  auto loadTiles = [&](int k0) {
    if (AMODE == 0) {
#pragma unroll
      for (int t = 0; t < 4; ++t)
        fa[t] = *(const f32x4*)(A32 + (size_t)(bm * BM + ra + 32 * t) * Kk + k0 + c4);
      // prefetch the tile after next on the heavy A stream (global_prefetch_b8)
      if (k0 + BK < Kk)
        __builtin_prefetch((const void*)(A32 + (size_t)(bm * BM + ra) * Kk + k0 + BK + c4), 0, 0);
    } else {
#pragma unroll
      for (int t = 0; t < 2; ++t)
        a16[t] = *(const u32x4*)(A16 + (size_t)(bm * BM + rb + 64 * t) * Kk + k0 + c8);
    }
#pragma unroll
    for (int t = 0; t < 2; ++t)
      breg[t] = *(const u32x4*)(B16 + (size_t)(bn * BN + rb + 64 * t) * Kk + k0 + c8);
  };

  auto storeTiles = [&](int buf) {
    if (AMODE == 0) {
#pragma unroll
      for (int t = 0; t < 4; ++t) {
        Pack4 p;
        p.h[0] = (bf16)fa[t].x; p.h[1] = (bf16)fa[t].y;
        p.h[2] = (bf16)fa[t].z; p.h[3] = (bf16)fa[t].w;
        *(u32x2*)(&As[buf][(ra + 32 * t) * LDT + c4]) = p.u;
      }
    } else {
#pragma unroll
      for (int t = 0; t < 2; ++t)
        *(u32x4*)(&As[buf][(rb + 64 * t) * LDT + c8]) = a16[t];
    }
#pragma unroll
    for (int t = 0; t < 2; ++t)
      *(u32x4*)(&Bs[buf][(rb + 64 * t) * LDT + c8]) = breg[t];
  };

  // prologue: fill buffer 0
  loadTiles(0);
  storeTiles(0);
  __syncthreads();

  int cur = 0;
  for (int k0 = 0; k0 < Kk; k0 += BK) {
    const bool more = (k0 + BK) < Kk;
    if (more) loadTiles(k0 + BK);  // global loads in flight during compute

    FragBF afr[4], bfr[2];
#pragma unroll
    for (int i = 0; i < 4; ++i) {
      const bf16* p = &As[cur][(wm + i * 16 + lanelo) * LDT + klo];
      afr[i].u[0] = *(const u32x4*)(p);
      afr[i].u[1] = *(const u32x4*)(p + 16);
    }
#pragma unroll
    for (int j = 0; j < 2; ++j) {
      const bf16* p = &Bs[cur][(wn + j * 16 + lanelo) * LDT + klo];
      bfr[j].u[0] = *(const u32x4*)(p);
      bfr[j].u[1] = *(const u32x4*)(p + 16);
    }
#pragma unroll
    for (int i = 0; i < 4; ++i)
#pragma unroll
      for (int j = 0; j < 2; ++j)
        acc[i][j] = __builtin_amdgcn_wmma_f32_16x16x32_bf16(
            false, afr[i].v, false, bfr[j].v, (short)0, acc[i][j], false, false);

    if (more) storeTiles(cur ^ 1);  // writes other buffer; no race with reads
    __syncthreads();
    cur ^= 1;
  }

  // ---- epilogue (C/D layout: lanes 0-15 -> M+0, lanes 16-31 -> M+8) ----
  float* Cf = (float*)Cg + (size_t)bz * sC;
  bf16*  Cb = (bf16*)Cg + (size_t)bz * sC;
  const int mhi = (lane >> 4) * 8;

#pragma unroll
  for (int i = 0; i < 4; ++i) {
#pragma unroll
    for (int j = 0; j < 2; ++j) {
      int gn  = bn * BN + wn + j * 16 + lanelo;
      int gm0 = bm * BM + wm + i * 16 + mhi;
      if (EMODE == 5) {
        // transposed packed store: ht[b][gn][gm%512], 8 consecutive M per lane
        Pack8 p;
#pragma unroll
        for (int v = 0; v < 8; ++v)
          p.h[v] = (bf16)fmaxf(acc[i][j][v] + ep0[gn], 0.f);
        int bb = gm0 >> 9;        // batch (512 rows per batch)
        int ml = gm0 & (N_ - 1);  // node index inside batch (8-aligned)
        *(u32x4*)((bf16*)Cg + (size_t)bb * HID_ * N_ + (size_t)gn * N_ + ml) = p.u;
      } else {
#pragma unroll
        for (int v = 0; v < 8; ++v) {
          int gm = gm0 + v;
          float val = acc[i][j][v];
          if (EMODE == 0) {
            int bb = gm >> 9;  // row -> batch (N_=512 rows per batch)
            float t = val + ep0[bb * HID_ + gn] + ep3[gm] * ep1[gn];
            Cf[(size_t)gm * Nn + gn] = fmaxf(t, 0.f) * ep2[gn];
          } else if (EMODE == 2) {
            Cf[(size_t)gm * Nn + gn] = val;
          } else if (EMODE == 3) {
            Cb[(size_t)gm * Nn + gn] = (bf16)val;
          } else {
            Cf[(size_t)gm * Nn + gn] = fmaxf(val + ep0[gn], 0.f);
          }
        }
      }
    }
  }
}

// ---------------- row norm: xn = xw / (||xw|| + 1e-8), bf16 out ----------------
__global__ void rownorm_kernel(const float* __restrict__ xw, bf16* __restrict__ xn) {
  int row  = blockIdx.x * 8 + (threadIdx.x >> 5);
  int lane = threadIdx.x & 31;
  const float* p = xw + (size_t)row * HID_;
  float vals[16];
  float ss = 0.f;
#pragma unroll
  for (int i = 0; i < 16; ++i) {
    float x = p[i * 32 + lane];
    vals[i] = x;
    ss += x * x;
  }
#pragma unroll
  for (int o = 16; o >= 1; o >>= 1) ss += __shfl_xor(ss, o, 32);
  float r = 1.0f / (sqrtf(ss) + 1e-8f);
  bf16* q = xn + (size_t)row * HID_;
#pragma unroll
  for (int i = 0; i < 16; ++i) q[i * 32 + lane] = (bf16)(vals[i] * r);
}

// ---------------- top-k threshold + eps + self loop + row normalize ----------------
__global__ void build_adj_kernel(const float* __restrict__ S, bf16* __restrict__ A) {
  int row  = blockIdx.x * 8 + (threadIdx.x >> 5);  // flat row in [0, B*N)
  int n    = row & (N_ - 1);
  int lane = threadIdx.x & 31;
  const float* p = S + (size_t)row * N_;
  float orig[16], work[16];
#pragma unroll
  for (int i = 0; i < 16; ++i) {
    float x = p[i * 32 + lane];
    orig[i] = x;
    work[i] = x;
  }
  float kth = 0.f;
  for (int it = 0; it < KTOP; ++it) {
    float bv = -3.4e38f;
    int bi = lane;
#pragma unroll
    for (int i = 0; i < 16; ++i) {
      if (work[i] > bv) { bv = work[i]; bi = i * 32 + lane; }
    }
#pragma unroll
    for (int o = 1; o < 32; o <<= 1) {  // (max, min-idx) butterfly reduce
      float ov = __shfl_xor(bv, o, 32);
      int   oi = __shfl_xor(bi, o, 32);
      if (ov > bv || (ov == bv && oi < bi)) { bv = ov; bi = oi; }
    }
    kth = bv;
    int slot = bi >> 5;
    if ((bi & 31) == lane) {
#pragma unroll
      for (int i = 0; i < 16; ++i)
        if (i == slot) work[i] = -3.4e38f;  // remove exactly one instance
    }
  }
  float a[16];
  float sum = 0.f;
#pragma unroll
  for (int i = 0; i < 16; ++i) {
    int j = i * 32 + lane;
    float s = orig[i];
    float v = (s >= kth && s > EPSV) ? s : 0.f;
    if (j == n) v += 1.f;  // self loop
    a[i] = v;
    sum += v;
  }
#pragma unroll
  for (int o = 16; o >= 1; o >>= 1) sum += __shfl_xor(sum, o, 32);
  float rs = 1.0f / sum;
  bf16* q = A + (size_t)row * N_;
#pragma unroll
  for (int i = 0; i < 16; ++i) q[i * 32 + lane] = (bf16)(a[i] * rs);
}

// ---------------- launch ----------------
extern "C" void kernel_launch(void* const* d_in, const int* in_sizes, int n_in,
                              void* d_out, int out_size, void* d_ws, size_t ws_size,
                              hipStream_t stream) {
  const float* ppl    = (const float*)d_in[0];
  const float* vhint  = (const float*)d_in[1];
  const float* answer = (const float*)d_in[2];
  const float* W_vh   = (const float*)d_in[3];
  const float* b_vh   = (const float*)d_in[4];
  const float* W_dh   = (const float*)d_in[5];
  const float* b_dh   = (const float*)d_in[6];
  const float* w_topo = (const float*)d_in[7];
  const float* W_in   = (const float*)d_in[8];
  const float* b_in   = (const float*)d_in[9];
  const float* W_gcn  = (const float*)d_in[10];
  const float* b_gcn  = (const float*)d_in[11];

  // workspace layout (needs ~120 MB; big buffers overlaid where lifetimes allow)
  char* ws = (char*)d_ws;
  float* u      = (float*)(ws);                  // [512]
  float* cvec   = (float*)(ws + 2048);           // [512]
  float* biasBJ = (float*)(ws + 4096);           // [32,512]
  bf16* WdhT = (bf16*)(ws + (128ull << 10));     // [512,2048]  (W_dh_ppl^T)
  bf16* WinT = WdhT + (size_t)DPPL * HID_;       // [512,2048]  (W_in^T)
  bf16* WgT  = WinT + (size_t)DPPL * HID_;       // [3,512,512] (W_gcn^T per layer)
  float* xw   = (float*)(ws + (8ull << 20));     // [16384,512] f32 (32MB)
  float* S    = xw;                              // S overlays xw (xw dead after rownorm)
  bf16* xn    = (bf16*)(ws + (40ull << 20));     // [16384,512] bf16
  bf16* Aadj  = (bf16*)(ws + (56ull << 20));     // [32,512,512] bf16
  bf16* ht0   = (bf16*)(ws + (72ull << 20));     // [32,512hid,512node] bf16 (transposed h)
  bf16* ht1   = (bf16*)(ws + (88ull << 20));     // [32,512,512] bf16
  bf16* msg   = (bf16*)(ws + (104ull << 20));    // [16384,512] bf16 row-major -> ends 120MB

  // rank-1 vh path + fused bias
  prep_uc_kernel<<<1, HID_, 0, stream>>>(W_vh, b_vh, W_dh, u, cvec);
  prep_bias_kernel<<<B_, HID_, 0, stream>>>(answer, W_dh, b_dh, cvec, biasBJ);

  // transpose+convert all weights once -> every GEMM B-stage is a contiguous row copy
  cvt_transpose_kernel<<<dim3(DPPL / 32, HID_ / 32, 1), 256, 0, stream>>>(W_dh, WdhT, DPPL, HID_);
  cvt_transpose_kernel<<<dim3(DPPL / 32, HID_ / 32, 1), 256, 0, stream>>>(W_in, WinT, DPPL, HID_);
  cvt_transpose_kernel<<<dim3(HID_ / 32, HID_ / 32, NLAYER), 256, 0, stream>>>(W_gcn, WgT, HID_, HID_);

  const int Mflat = B_ * N_;  // 16384

  // xw = relu(ppl@Wdh_ppl + bias[b,:] + vh*u) * w_topo   (f32, row-major)
  gemm_wmma_kernel<0, 0><<<dim3(Mflat / BM, HID_ / BN, 1), 256, 0, stream>>>(
      ppl, WdhT, xw, Mflat, HID_, DPPL, 0, 0, 0, biasBJ, u, w_topo, vhint);

  rownorm_kernel<<<Mflat / 8, 256, 0, stream>>>(xw, xn);

  // S[b] = xn[b] @ xn[b]^T  (B operand = xn rows, already K-major)
  gemm_wmma_kernel<1, 2><<<dim3(N_ / BM, N_ / BN, B_), 256, 0, stream>>>(
      xn, xn, S, N_, N_, HID_, (long)N_ * HID_, (long)N_ * HID_, (long)N_ * N_,
      nullptr, nullptr, nullptr, nullptr);

  build_adj_kernel<<<Mflat / 8, 256, 0, stream>>>(S, Aadj);

  // ht0 = relu(ppl @ W_in + b_in)^T  (transposed packed bf16 store)
  gemm_wmma_kernel<0, 5><<<dim3(Mflat / BM, HID_ / BN, 1), 256, 0, stream>>>(
      ppl, WinT, ht0, Mflat, HID_, DPPL, 0, 0, 0, b_in, nullptr, nullptr, nullptr);

  bf16* cur = ht0;
  bf16* nxt = ht1;
  for (int step = 0; step < STEPS_; ++step) {
    int l = step % NLAYER;
    // msg[b] = A[b] @ h[b] : B operand is ht[b] (K-major by construction) -> row copies
    gemm_wmma_kernel<1, 3><<<dim3(N_ / BM, N_ / BN, B_), 256, 0, stream>>>(
        Aadj, cur, msg, N_, N_, N_, (long)N_ * N_, (long)N_ * HID_, (long)N_ * HID_,
        nullptr, nullptr, nullptr, nullptr);
    if (step < STEPS_ - 1) {
      // h^T = relu(msg @ W_l + b_l)^T   (B operand = WgT, output transposed packed)
      gemm_wmma_kernel<1, 5><<<dim3(Mflat / BM, HID_ / BN, 1), 256, 0, stream>>>(
          msg, WgT + (size_t)l * HID_ * HID_, nxt, Mflat, HID_, HID_, 0, 0, 0,
          b_gcn + l * HID_, nullptr, nullptr, nullptr);
      bf16* t = cur; cur = nxt; nxt = t;
    } else {
      // final step writes f32 row-major straight to d_out
      gemm_wmma_kernel<1, 4><<<dim3(Mflat / BM, HID_ / BN, 1), 256, 0, stream>>>(
          msg, WgT + (size_t)l * HID_ * HID_, d_out, Mflat, HID_, HID_, 0, 0, 0,
          b_gcn + l * HID_, nullptr, nullptr, nullptr);
    }
  }
}
